// FP4SymmetricLinear_22093311771209
// MI455X (gfx1250) — compile-verified, matrix operand order checked by reading, not run on hardware
//
#include <hip/hip_runtime.h>
#include <hip/hip_bf16.h>

typedef __attribute__((ext_vector_type(16))) __bf16 v16bf;
typedef __attribute__((ext_vector_type(2)))  __bf16 bf16x2;
typedef __attribute__((ext_vector_type(8)))  float  v8f;

#define M_TOTAL 8192
#define N_TOTAL 4096
#define K_TOTAL 4096
#define BM 128
#define BN 128
#define BK 32
#define KTILES (K_TOTAL / BK)
#define LDS_STRIDE 40   // 32 + 8 shorts pad -> 80B rows (mult of 16B), conflict-free b128

__device__ __constant__ float FP4_CODEBOOK[16] = {
    0.0f, 0.0052f, 0.6667f, 1.0f, 0.3333f, 0.5f, 0.1667f, 0.25f,
    0.0f, -0.0052f, -0.6667f, -1.0f, -0.3333f, -0.5f, -0.1667f, -0.25f};

__device__ __forceinline__ unsigned rne_bits(float f) {
    union { float f; unsigned u; } x; x.f = f;
    return x.u + 0x7FFFu + ((x.u >> 16) & 1u);   // bf16 RNE in bits [31:16]
}

// Two f32 -> packed bf16 pair: 2x (bfe+add3) + 1x v_perm_b32.
__device__ __forceinline__ unsigned pack_bf16(float a, float b) {
    unsigned ra = rne_bits(a);
    unsigned rb = rne_bits(b);
    // dst[15:0] = ra[31:16] (src1 bytes 2,3), dst[31:16] = rb[31:16] (src0 bytes 6,7)
    return __builtin_amdgcn_perm(rb, ra, 0x07060302u);
}

__device__ __forceinline__ v16bf load_frag(const unsigned short* rowp, int col) {
    union { uint4 q[2]; v16bf v; } u;
    u.q[0] = *(const uint4*)(rowp + col);       // K = h*8 .. h*8+7
    u.q[1] = *(const uint4*)(rowp + col + 16);  // K = 16+h*8 .. +7
    return u.v;
}

__global__ __launch_bounds__(256)
void fp4_linear_wmma_kernel(const float* __restrict__ x,
                            const int* __restrict__ wpk,
                            const float* __restrict__ scale,
                            const float* __restrict__ bias,
                            float* __restrict__ out) {
    // Double-buffered tiles: one barrier per K-step.
    __shared__ __align__(16) unsigned short As[2][BM][LDS_STRIDE];
    __shared__ __align__(16) unsigned short Bs[2][BN][LDS_STRIDE];
    __shared__ unsigned lutp[256];  // byte -> packed (bf16(cb[high]), bf16(cb[low]))

    const int tid   = threadIdx.x;
    const int lane  = tid & 31;
    const int wave  = tid >> 5;
    const int waveM = wave & 3;   // 4 waves along M
    const int waveN = wave >> 2;  // 2 waves along N
    const int h     = lane >> 4;
    const int l15   = lane & 15;

    const int blockN = blockIdx.x * BN;
    const int blockM = blockIdx.y * BM;

    // Byte -> bf16-pair LUT (RNE-rounded once, full precision source).
    lutp[tid & 255] = pack_bf16(FP4_CODEBOOK[(tid >> 4) & 15], FP4_CODEBOOK[tid & 15]);

    v8f acc[2][4];
#pragma unroll
    for (int im = 0; im < 2; ++im)
#pragma unroll
        for (int in = 0; in < 4; ++in)
            acc[im][in] = (v8f){0.f, 0.f, 0.f, 0.f, 0.f, 0.f, 0.f, 0.f};

    // W mapping: 2 threads per N-row, 8 packed bytes (16 K values) each
    const int wrow = tid >> 1;
    const int wh   = tid & 1;
    const int oglb = blockN + wrow;

    float4   aR[2][2];
    int4     wR[2];
    unsigned sPk;   // bf16(scale) broadcast pair

#define LOAD_TILES(K0)                                                                \
    do {                                                                              \
        _Pragma("unroll")                                                             \
        for (int u = 0; u < 2; ++u) {                                                 \
            int p = tid + u * 256;                                                    \
            int row = p >> 2, c8 = (p & 3) << 3;                                      \
            const float* gp = x + (size_t)(blockM + row) * K_TOTAL + (K0) + c8;       \
            aR[u][0] = *(const float4*)gp;                                            \
            aR[u][1] = *(const float4*)(gp + 4);                                      \
        }                                                                             \
        const int* pbase = wpk + (size_t)oglb * (K_TOTAL / 2) + ((K0) >> 1) + wh * 8; \
        wR[0] = *(const int4*)pbase;                                                  \
        wR[1] = *(const int4*)(pbase + 4);                                            \
        float s = scale[oglb * (K_TOTAL / 64) + ((K0) >> 6)];                         \
        sPk = pack_bf16(s, s);                                                        \
    } while (0)

#define STAGE(BUF)                                                                    \
    do {                                                                              \
        _Pragma("unroll")                                                             \
        for (int u = 0; u < 2; ++u) {                                                 \
            int p = tid + u * 256;                                                    \
            int row = p >> 2, c8 = (p & 3) << 3;                                      \
            uint4 st;                                                                 \
            st.x = pack_bf16(aR[u][0].x, aR[u][0].y);                                 \
            st.y = pack_bf16(aR[u][0].z, aR[u][0].w);                                 \
            st.z = pack_bf16(aR[u][1].x, aR[u][1].y);                                 \
            st.w = pack_bf16(aR[u][1].z, aR[u][1].w);                                 \
            *(uint4*)(&As[BUF][row][c8]) = st;                                        \
        }                                                                             \
        /* byte values are 0..255 by construction: no mask needed */                  \
        int pk[8] = {wR[0].x, wR[0].y, wR[0].z, wR[0].w,                              \
                     wR[1].x, wR[1].y, wR[1].z, wR[1].w};                             \
        union { unsigned u; bf16x2 v; } sb; sb.u = sPk;                               \
        unsigned o[8];                                                                \
        _Pragma("unroll")                                                             \
        for (int q = 0; q < 8; ++q) {                                                 \
            union { unsigned u; bf16x2 v; } l, r;                                     \
            l.u = lutp[pk[q]];                                                        \
            r.v = l.v * sb.v;            /* v_pk_mul_bf16 */                          \
            o[q] = r.u;                                                               \
        }                                                                             \
        const int kbase = wh * 16;                                                    \
        *(uint4*)(&Bs[BUF][wrow][kbase])     = make_uint4(o[0], o[1], o[2], o[3]);    \
        *(uint4*)(&Bs[BUF][wrow][kbase + 8]) = make_uint4(o[4], o[5], o[6], o[7]);    \
    } while (0)

    // Prologue: fill buffer 0, prefetch tile 1 into registers.
    LOAD_TILES(0);
    __syncthreads();   // lutp visible before first STAGE reads it
    STAGE(0);
    LOAD_TILES(BK);
    __syncthreads();

    for (int kt = 0; kt < KTILES; ++kt) {
        const int cur = kt & 1;

        // ---- fragments from current buffer ----
        v16bf aF[2], bF[4];
        const int col = h * 8;
#pragma unroll
        for (int im = 0; im < 2; ++im)
            aF[im] = load_frag(&As[cur][waveM * 32 + im * 16 + l15][0], col);
#pragma unroll
        for (int in = 0; in < 4; ++in)
            bF[in] = load_frag(&Bs[cur][waveN * 64 + in * 16 + l15][0], col);

        // ---- stage tile kt+1 (registers) into the other buffer ----
        if (kt + 1 < KTILES) STAGE(cur ^ 1);

        // ---- issue global loads for tile kt+2 ----
        if (kt + 2 < KTILES) LOAD_TILES((kt + 2) * BK);

        // ---- WMMA ----
#pragma unroll
        for (int im = 0; im < 2; ++im)
#pragma unroll
            for (int in = 0; in < 4; ++in)
                acc[im][in] = __builtin_amdgcn_wmma_f32_16x16x32_bf16(
                    false, aF[im], false, bF[in], (short)0, acc[im][in], false, false);

        __syncthreads();
    }
#undef LOAD_TILES
#undef STAGE

    // ---- epilogue: bias add + store (C/D layout: lane->N, reg r -> M = r + 8*h) ----
#pragma unroll
    for (int in = 0; in < 4; ++in) {
        const int coln = blockN + waveN * 64 + in * 16 + l15;
        const float bv = bias[coln];
#pragma unroll
        for (int im = 0; im < 2; ++im) {
            const int rowbase = blockM + waveM * 32 + im * 16 + 8 * h;
#pragma unroll
            for (int r = 0; r < 8; ++r) {
                out[(size_t)(rowbase + r) * N_TOTAL + coln] = acc[im][in][r] + bv;
            }
        }
    }
}

extern "C" void kernel_launch(void* const* d_in, const int* in_sizes, int n_in,
                              void* d_out, int out_size, void* d_ws, size_t ws_size,
                              hipStream_t stream) {
    const float* x     = (const float*)d_in[0];
    const int*   wpk   = (const int*)d_in[1];
    const float* scale = (const float*)d_in[2];
    const float* bias  = (const float*)d_in[3];
    float* out = (float*)d_out;

    dim3 grid(N_TOTAL / BN, M_TOTAL / BM);  // 32 x 64
    dim3 block(256);
    fp4_linear_wmma_kernel<<<grid, block, 0, stream>>>(x, wpk, scale, bias, out);
}
